// NGCFDL_60473139527900
// MI455X (gfx1250) — matrix operand order repaired and checked by reading, not compiled, not run on hardware
//
#include <hip/hip_runtime.h>
#include <math.h>

#define NUSERS 100000
#define NITEMS 100000
#define NNODES 200000
#define NEDGES 3200000
#define TXT    768
#define GREC   16

typedef __attribute__((ext_vector_type(16))) _Float16 v16h;
typedef __attribute__((ext_vector_type(8)))  _Float16 v8h;
typedef __attribute__((ext_vector_type(4)))  _Float16 v4h;
typedef __attribute__((ext_vector_type(8)))  float    v8f;

// wave-level LDS sync: same-wave LDS ops are in-order (ISA ch.11), so a single
// DScnt wait between the stores and the cross-lane loads is sufficient.
__device__ __forceinline__ void wave_lds_sync() {
  __builtin_amdgcn_wave_barrier();
  asm volatile("s_wait_dscnt 0x0" ::: "memory");
  __builtin_amdgcn_wave_barrier();
}

// v16h from two contiguous 16B chunks at p and p+8 halves
__device__ __forceinline__ v16h load_frag16(const _Float16* p) {
  v8h lo = *(const v8h*)p;
  v8h hi = *(const v8h*)(p + 8);
  v16h r;
#pragma unroll
  for (int i = 0; i < 8; ++i) { r[i] = lo[i]; r[8 + i] = hi[i]; }
  return r;
}

// A-fragment: two 16B chunks at p and p+16 halves (K gap of 8 between chunks)
__device__ __forceinline__ v16h load_frag_a(const _Float16* p) {
  v8h lo = *(const v8h*)p;
  v8h hi = *(const v8h*)(p + 16);
  v16h r;
#pragma unroll
  for (int i = 0; i < 8; ++i) { r[i] = lo[i]; r[8 + i] = hi[i]; }
  return r;
}

// ---------- Kernel 1: text MLP  ego[:,0:16]=grec, ego[:,16:32]=relu(X@W1)@W2 ----------
// block = 128 threads = 4 waves, each wave owns one 16-row tile.
__global__ __launch_bounds__(128)
void ngcf_mlp_kernel(const float* __restrict__ X,    // (nrows, 768)
                     const float* __restrict__ W1,   // (768, 32)
                     const float* __restrict__ W2,   // (32, 16)
                     const float* __restrict__ grec, // (nrows, 16)
                     float* __restrict__ ego,        // (NNODES, 32)
                     float* __restrict__ out,        // (NNODES, 128)
                     int nTiles, int rowOffset)
{
  // transposed weights: Wt[col][k] -> B-fragments are contiguous 32B per lane
  __shared__ __align__(16) _Float16 w1t[32][784];   // 768 + 16 pad (stride 1568B) ~49KB
  __shared__ __align__(16) _Float16 w2t[16][48];    // 32 + 16 pad
  __shared__ __align__(16) _Float16 slab[4][16 * 40]; // X slab f16, 80B row stride; reused for H

  const int tid = threadIdx.x;
  for (int i = tid; i < TXT * 32; i += 128) { const int k = i >> 5, cl = i & 31; w1t[cl][k] = (_Float16)W1[i]; }
  for (int i = tid; i < 32 * GREC;  i += 128) { const int k = i >> 4, cl = i & 15; w2t[cl][k] = (_Float16)W2[i]; }

  const int wave = tid >> 5;
  const int lane = tid & 31;
  int tileIdx = blockIdx.x * 4 + wave;
  const bool valid = tileIdx < nTiles;
  if (!valid) tileIdx = nTiles - 1;        // keep EXEC full through the WMMAs
  const int rowbase = tileIdx * 16;

  const int row  = lane & 15;
  const int kg8  = (lane >> 4) << 3;       // 0 / 8
  const int kg16 = (lane >> 4) << 4;       // 0 / 16
  const int rb   = (lane >> 4) << 3;
  const int c    = lane & 15;

  __syncthreads();

  _Float16* myslab = slab[wave];

  // GEMM1: H = relu(X_tile(16x768) @ W1(768x32)), K-loop in steps of 32
  v8f acc0 = {};
  v8f acc1 = {};
  for (int kb = 0; kb < TXT; kb += 32) {
    // coalesced stage of the 16x32 f32 slab -> f16 LDS (padded rows)
#pragma unroll
    for (int j = 0; j < 4; ++j) {
      const int ch = lane + 32 * j;              // 0..127 chunks of 4 floats
      const int r  = ch >> 3;
      const int c4 = (ch & 7) * 4;
      const float4 x = *(const float4*)(X + (size_t)(rowbase + r) * TXT + kb + c4);
      v4h hx; hx[0] = (_Float16)x.x; hx[1] = (_Float16)x.y; hx[2] = (_Float16)x.z; hx[3] = (_Float16)x.w;
      *(v4h*)(myslab + r * 40 + c4) = hx;
    }
    wave_lds_sync();

    const v16h a  = load_frag16(myslab + row * 40 + kg16);
    const v16h b0 = load_frag16(&w1t[c][kb + kg16]);
    const v16h b1 = load_frag16(&w1t[16 + c][kb + kg16]);
    acc0 = __builtin_amdgcn_wmma_f32_16x16x32_f16(false, a, false, b0, (short)0, acc0, false, false);
    acc1 = __builtin_amdgcn_wmma_f32_16x16x32_f16(false, a, false, b1, (short)0, acc1, false, false);
  }

  // relu + restage H (D-layout -> padded row-major f16, reusing the slab)
#pragma unroll
  for (int r = 0; r < 8; ++r) {
    myslab[(rb + r) * 40 + c]      = (_Float16)fmaxf(acc0[r], 0.0f);
    myslab[(rb + r) * 40 + 16 + c] = (_Float16)fmaxf(acc1[r], 0.0f);
  }
  wave_lds_sync();

  // GEMM2: created = H(16x32) @ W2(32x16), single WMMA
  const v16h aH  = load_frag16(myslab + row * 40 + kg16);
  const v16h bW2 = load_frag16(&w2t[c][kg16]);
  v8f accC = {};
  accC = __builtin_amdgcn_wmma_f32_16x16x32_f16(false, aH, false, bW2, (short)0, accC, false, false);

  if (valid) {
#pragma unroll
    for (int r = 0; r < 8; ++r) {
      const int lrow = rowbase + rb + r;          // row within this half (user or item)
      const int node = rowOffset + lrow;          // global node id
      const float g  = grec[(size_t)lrow * GREC + c];
      const float cr = accC[r];
      ego[(size_t)node * 32 + c]       = g;
      ego[(size_t)node * 32 + 16 + c]  = cr;
      out[(size_t)node * 128 + c]      = g;       // all_embeddings[0] (unnormalized)
      out[(size_t)node * 128 + 16 + c] = cr;
    }
  }
}

// ---------- Kernel 2: edge gather + segment-sum via f32 atomics ----------
// 8 threads per edge, each handles one float4 of the 32-dim row.
__global__ __launch_bounds__(256)
void ngcf_scatter_kernel(const int* __restrict__ erow, const int* __restrict__ ecol,
                         const float* __restrict__ vals,
                         const float* __restrict__ ego, float* __restrict__ side, int E)
{
  const int t = blockIdx.x * blockDim.x + threadIdx.x;
  const int e = t >> 3;
  if (e >= E) return;
  const int part = t & 7;
  const int src = ecol[e];
  const int dst = erow[e];
  const float v = vals[e];
  const float4 g = ((const float4*)(ego + (size_t)src * 32))[part];
  float* d = side + (size_t)dst * 32 + part * 4;
  atomicAdd(d + 0, v * g.x);
  atomicAdd(d + 1, v * g.y);
  atomicAdd(d + 2, v * g.z);
  atomicAdd(d + 3, v * g.w);
}

// ---------- Kernel 3: per-layer transform + row-normalize ----------
// ego_new = lrelu(side@gcW + gcb) + lrelu((ego*side)@biW + bib); out slice = normalize(ego_new)
__global__ __launch_bounds__(128)
void ngcf_layer_kernel(float* __restrict__ ego,         // (NNODES,32), updated in place
                       const float* __restrict__ side,  // (NNODES,32)
                       const float* __restrict__ gcW, const float* __restrict__ gcb,
                       const float* __restrict__ biW, const float* __restrict__ bib,
                       float* __restrict__ out, int colOffset)
{
  __shared__ __align__(16) _Float16 gwt[32][48];       // transposed gc_W
  __shared__ __align__(16) _Float16 bwt[32][48];       // transposed bi_W
  __shared__ __align__(16) float sS[4][16 * 36];       // side slab, 144B row stride
  __shared__ __align__(16) float sE[4][16 * 36];       // ego slab

  const int tid = threadIdx.x;
  for (int i = tid; i < 32 * 32; i += 128) {
    const int k = i >> 5, cl = i & 31;
    gwt[cl][k] = (_Float16)gcW[i];
    bwt[cl][k] = (_Float16)biW[i];
  }

  const int wave = tid >> 5;
  const int lane = tid & 31;
  const int tile = blockIdx.x * 4 + wave;   // 12500 tiles = 3125 blocks * 4 waves, exact
  const int rowbase = tile * 16;

  // async copy of the 16x32 f32 side/ego tiles into per-wave LDS (CDNA5 async path)
  {
    const unsigned ldsS = (unsigned)(uintptr_t)sS[wave];
    const unsigned ldsE = (unsigned)(uintptr_t)sE[wave];
    const float* gs = side + (size_t)rowbase * 32;
    const float* ge = ego  + (size_t)rowbase * 32;
#pragma unroll
    for (int j = 0; j < 4; ++j) {
      const int ch = lane + 32 * j;
      const int r  = ch >> 3;
      const int c4 = (ch & 7) * 4;
      const unsigned loff = (unsigned)((r * 36 + c4) * 4);
      asm volatile("global_load_async_to_lds_b128 %0, %1, off"
                   :: "v"(ldsS + loff),
                      "v"((unsigned long long)(uintptr_t)(gs + r * 32 + c4))
                   : "memory");
      asm volatile("global_load_async_to_lds_b128 %0, %1, off"
                   :: "v"(ldsE + loff),
                      "v"((unsigned long long)(uintptr_t)(ge + r * 32 + c4))
                   : "memory");
    }
  }
  __syncthreads();                               // weights staged by whole block
  asm volatile("s_wait_asynccnt 0x0" ::: "memory");
  __builtin_amdgcn_wave_barrier();

  // A fragments: side and ego*side (conflict-free scattered f32 LDS reads)
  const int row  = lane & 15;
  const int kg8  = (lane >> 4) << 3;
  const int kg16 = (lane >> 4) << 4;
  v16h aS, aP;
#pragma unroll
  for (int h = 0; h < 16; ++h) {
    const int k = kg16 + h;
    const float s = sS[wave][row * 36 + k];
    const float e = sE[wave][row * 36 + k];
    aS[h] = (_Float16)s;
    aP[h] = (_Float16)(s * e);
  }
  const int c = lane & 15;
  const v16h bG0 = load_frag16(&gwt[c][kg16]);
  const v16h bG1 = load_frag16(&gwt[16 + c][kg16]);
  const v16h bB0 = load_frag16(&bwt[c][kg16]);
  const v16h bB1 = load_frag16(&bwt[16 + c][kg16]);

  // bias folded into accumulator init (same bias for every row of a column)
  const float g0b = gcb[c], g1b = gcb[16 + c], b0b = bib[c], b1b = bib[16 + c];
  v8f cG0, cG1, cB0, cB1;
#pragma unroll
  for (int r = 0; r < 8; ++r) { cG0[r] = g0b; cG1[r] = g1b; cB0[r] = b0b; cB1[r] = b1b; }

  cG0 = __builtin_amdgcn_wmma_f32_16x16x32_f16(false, aS, false, bG0, (short)0, cG0, false, false);
  cG1 = __builtin_amdgcn_wmma_f32_16x16x32_f16(false, aS, false, bG1, (short)0, cG1, false, false);
  cB0 = __builtin_amdgcn_wmma_f32_16x16x32_f16(false, aP, false, bB0, (short)0, cB0, false, false);
  cB1 = __builtin_amdgcn_wmma_f32_16x16x32_f16(false, aP, false, bB1, (short)0, cB1, false, false);

  // leaky_relu (slope 0.01), sum, per-row L2 norm via shfl_xor inside 16-lane halves
  const int rb = (lane >> 4) << 3;
  float e0[8], e1[8];
#pragma unroll
  for (int r = 0; r < 8; ++r) {
    float g0 = cG0[r]; g0 = g0 > 0.0f ? g0 : 0.01f * g0;
    float g1 = cG1[r]; g1 = g1 > 0.0f ? g1 : 0.01f * g1;
    float b0 = cB0[r]; b0 = b0 > 0.0f ? b0 : 0.01f * b0;
    float b1 = cB1[r]; b1 = b1 > 0.0f ? b1 : 0.01f * b1;
    e0[r] = g0 + b0;
    e1[r] = g1 + b1;
  }
#pragma unroll
  for (int r = 0; r < 8; ++r) {
    float ss = e0[r] * e0[r] + e1[r] * e1[r];
    ss += __shfl_xor(ss, 1);
    ss += __shfl_xor(ss, 2);
    ss += __shfl_xor(ss, 4);
    ss += __shfl_xor(ss, 8);           // masks <16 stay within each 16-lane half = one row
    const float scale = 1.0f / fmaxf(sqrtf(ss), 1e-12f);
    const int node = rowbase + rb + r;
    ego[(size_t)node * 32 + c]                   = e0[r];
    ego[(size_t)node * 32 + 16 + c]              = e1[r];
    out[(size_t)node * 128 + colOffset + c]      = e0[r] * scale;
    out[(size_t)node * 128 + colOffset + 16 + c] = e1[r] * scale;
  }
}

// ---------- launcher ----------
extern "C" void kernel_launch(void* const* d_in, const int* in_sizes, int n_in,
                              void* d_out, int out_size, void* d_ws, size_t ws_size,
                              hipStream_t stream) {
  const int*   edge_row = (const int*)  d_in[0];
  const int*   edge_col = (const int*)  d_in[1];
  const float* adj_vals = (const float*)d_in[2];
  const float* uname    = (const float*)d_in[3];
  const float* sent     = (const float*)d_in[4];
  const float* uemb     = (const float*)d_in[5];
  const float* iemb     = (const float*)d_in[6];
  const float* ueW1     = (const float*)d_in[7];
  const float* ueW2     = (const float*)d_in[8];
  const float* ieW1     = (const float*)d_in[9];
  const float* ieW2     = (const float*)d_in[10];
  const float* gcW      = (const float*)d_in[11];
  const float* gcb      = (const float*)d_in[12];
  const float* biW      = (const float*)d_in[13];
  const float* bib      = (const float*)d_in[14];
  float* out = (float*)d_out;

  float* ego  = (float*)d_ws;                    // 200000*32 f32 = 25.6 MB
  float* side = ego + (size_t)NNODES * 32;       // 25.6 MB

  const int uTiles = NUSERS / 16;                // 6250
  const int iTiles = NITEMS / 16;                // 6250
  ngcf_mlp_kernel<<<(uTiles + 3) / 4, 128, 0, stream>>>(uname, ueW1, ueW2, uemb, ego, out, uTiles, 0);
  ngcf_mlp_kernel<<<(iTiles + 3) / 4, 128, 0, stream>>>(sent,  ieW1, ieW2, iemb, ego, out, iTiles, NUSERS);

  for (int l = 0; l < 3; ++l) {
    hipMemsetAsync(side, 0, (size_t)NNODES * 32 * sizeof(float), stream);
    const long long work = (long long)NEDGES * 8;
    ngcf_scatter_kernel<<<(int)((work + 255) / 256), 256, 0, stream>>>(edge_row, edge_col, adj_vals, ego, side, NEDGES);
    ngcf_layer_kernel<<<(NNODES / 16) / 4, 128, 0, stream>>>(ego, side,
        gcW + l * 32 * 32, gcb + l * 32, biW + l * 32 * 32, bib + l * 32,
        out, 32 * (l + 1));
  }
}
